// SLUTagging_438086664304
// MI455X (gfx1250) — matrix operand order repaired and checked by reading, not compiled, not run on hardware
//
#include <hip/hip_runtime.h>
#include <math.h>

#define Tn 512
#define Mm 512
#define Dd 128
#define Ee 512
#define NT 74
#define GHh 64
#define LHh 256

typedef __attribute__((ext_vector_type(16))) _Float16 v16h;
typedef __attribute__((ext_vector_type(8)))  float    v8f;

__device__ __forceinline__ float sigmoidf_(float x) { return 1.0f / (1.0f + expf(-x)); }

// ---------------------------------------------------------------------------
// WMMA GEMM: C[M x N] = A[M x K] @ W[N x K]^T + bias[N]
// 4 waves per block (one 16-row M-tile each); each wave computes a 16x32
// output strip with two v_wmma_f32_16x16x32_f16 accumulators sharing one A
// fragment per K-chunk. All loads are branchless: out-of-range N columns
// clamp the W row pointer to row 0 (valid, finite) and the garbage only
// lands in accumulator columns the guarded store never writes.
// Register layouts used by this code (wave32; A/B use contiguous k, not the ISA table order):
//   A: lane l holds row M=l%16; elem i -> K=k0+16g+i (16 contiguous floats)
//   B: lane l holds W-row N=l%16; elem i -> K=k0+16g+i (16 contiguous floats)
//   C/D: vgpr r, lane l -> (M = r + 8*(l/16), N = l%16)
// ---------------------------------------------------------------------------
__global__ void gemm_wmma_bias(const float* __restrict__ A,
                               const float* __restrict__ W,
                               const float* __restrict__ bias,
                               float* __restrict__ C,
                               int N, int K) {
  const int lane = threadIdx.x & 31;
  const int wave = threadIdx.x >> 5;
  const int g    = lane >> 4;
  const int l15  = lane & 15;
  const int m0   = (blockIdx.y * 4 + wave) * 16;
  const int n0   = blockIdx.x * 32;
  const int nr0  = n0 + l15;
  const int nr1  = n0 + 16 + l15;

  const float* arow  = A + (size_t)(m0 + l15) * K + 16 * g;
  const float* brow0 = W + (size_t)(nr0 < N ? nr0 : 0) * K + 16 * g;
  const float* brow1 = W + (size_t)(nr1 < N ? nr1 : 0) * K + 16 * g;

  v8f acc0 = {}, acc1 = {};
  for (int k0 = 0; k0 < K; k0 += 32) {
    v16h a, b0, b1;
#pragma unroll
    for (int i = 0; i < 8; ++i) {
      a[i]     = (_Float16)arow[k0 + i];
      a[8 + i] = (_Float16)arow[k0 + 8 + i];
    }
#pragma unroll
    for (int i = 0; i < 16; ++i) {
      b0[i] = (_Float16)brow0[k0 + i];
      b1[i] = (_Float16)brow1[k0 + i];
    }
    acc0 = __builtin_amdgcn_wmma_f32_16x16x32_f16(false, a, false, b0,
                                                  (short)0, acc0, false, false);
    acc1 = __builtin_amdgcn_wmma_f32_16x16x32_f16(false, a, false, b1,
                                                  (short)0, acc1, false, false);
  }

  if (nr0 < N) {
    const float bv = bias[nr0];
#pragma unroll
    for (int r = 0; r < 8; ++r)
      C[(size_t)(m0 + r + 8 * g) * N + nr0] = acc0[r] + bv;
  }
  if (nr1 < N) {
    const float bv = bias[nr1];
#pragma unroll
    for (int r = 0; r < 8; ++r)
      C[(size_t)(m0 + r + 8 * g) * N + nr1] = acc1[r] + bv;
  }
}

// ---------------------------------------------------------------------------
// scores[m] = <memory[m,:,:], u>   (one block per memory, float4 streaming)
// ---------------------------------------------------------------------------
__global__ void mem_dot_kernel(const float* __restrict__ mem,
                               const float* __restrict__ u,
                               float* __restrict__ scores) {
  __shared__ float red[256];
  const int m = blockIdx.x;
  const float4* m4 = (const float4*)(mem + (size_t)m * Tn * Dd);
  const float4* u4 = (const float4*)u;
  float acc = 0.f;
  for (int i = threadIdx.x; i < (Tn * Dd) / 4; i += 256) {
    float4 a = m4[i], b = u4[i];
    acc += a.x * b.x + a.y * b.y + a.z * b.z + a.w * b.w;
  }
  red[threadIdx.x] = acc;
  __syncthreads();
  for (int s = 128; s > 0; s >>= 1) {
    if (threadIdx.x < s) red[threadIdx.x] += red[threadIdx.x + s];
    __syncthreads();
  }
  if (threadIdx.x == 0) scores[m] = red[0];
}

__global__ void softmax512_kernel(const float* __restrict__ s, float* __restrict__ p) {
  __shared__ float red[512];
  const int t = threadIdx.x;
  float v = s[t];
  red[t] = v; __syncthreads();
  for (int k = 256; k > 0; k >>= 1) { if (t < k) red[t] = fmaxf(red[t], red[t + k]); __syncthreads(); }
  float mx = red[0]; __syncthreads();
  float e = expf(v - mx);
  red[t] = e; __syncthreads();
  for (int k = 256; k > 0; k >>= 1) { if (t < k) red[t] += red[t + k]; __syncthreads(); }
  p[t] = e / red[0];
}

// mbar[t,d] = sum_m p[m]*memory[m,t,d]   (one block per t, p cached in LDS)
__global__ void mem_wsum_kernel(const float* __restrict__ mem,
                                const float* __restrict__ p,
                                float* __restrict__ mbar) {
  __shared__ float pl[Mm];
  const int t = blockIdx.x;
  for (int i = threadIdx.x; i < Mm; i += blockDim.x) pl[i] = p[i];
  __syncthreads();
  const int d = threadIdx.x;
  float acc = 0.f;
  for (int m = 0; m < Mm; ++m) acc += pl[m] * mem[((size_t)m * Tn + t) * Dd + d];
  mbar[t * Dd + d] = acc;
}

__global__ void concat_kernel(const float* __restrict__ mbar,
                              const float* __restrict__ u,
                              float* __restrict__ cat) {
  const int idx = blockIdx.x * 256 + threadIdx.x;  // Tn * 2D total
  const int t = idx >> 8, j = idx & 255;
  cat[idx] = (j < Dd) ? mbar[t * Dd + j] : u[t * Dd + (j - Dd)];
}

__global__ void add_kernel(const float* __restrict__ a, const float* __restrict__ b,
                           float* __restrict__ c, int n) {
  const int i = blockIdx.x * 256 + threadIdx.x;
  if (i < n) c[i] = a[i] + b[i];
}

__global__ void transpose_kernel(const float* __restrict__ src, float* __restrict__ dst,
                                 int rows, int cols) {
  const int idx = blockIdx.x * 256 + threadIdx.x;
  if (idx < rows * cols) {
    const int r = idx / cols, c = idx % cols;
    dst[(size_t)c * rows + r] = src[idx];
  }
}

// ---------------------------------------------------------------------------
// BiGRU scan: block 0 = forward, block 1 = backward. 192 threads (6 waves).
// Whh (48KB, padded stride 65 vs 64-bank LDS) cached in LDS. PyTorch r,z,n order.
// ---------------------------------------------------------------------------
__global__ void gru_scan_kernel(const float* __restrict__ giF, const float* __restrict__ giB,
                                const float* __restrict__ WhhF, const float* __restrict__ WhhB,
                                const float* __restrict__ bhhF, const float* __restrict__ bhhB,
                                float* __restrict__ hout) {
  __shared__ float sW[3 * GHh * (GHh + 1)];
  __shared__ float sb[3 * GHh];
  __shared__ float sh[GHh];
  __shared__ float sg[3 * GHh];
  const int dir = blockIdx.x;
  const float* gi  = dir ? giB  : giF;
  const float* Whh = dir ? WhhB : WhhF;
  const float* bhh = dir ? bhhB : bhhF;
  const int tid = threadIdx.x;  // 0..191
  for (int i = tid; i < 3 * GHh * GHh; i += 3 * GHh)
    sW[(i / GHh) * (GHh + 1) + (i % GHh)] = Whh[i];
  if (tid < 3 * GHh) sb[tid] = bhh[tid];
  if (tid < GHh) sh[tid] = 0.f;
  __syncthreads();
  for (int s = 0; s < Tn; ++s) {
    const int t = dir ? (Tn - 1 - s) : s;
    float acc = sb[tid];
    const float* wr = &sW[tid * (GHh + 1)];
#pragma unroll 8
    for (int k = 0; k < GHh; ++k) acc += sh[k] * wr[k];
    sg[tid] = acc;
    __syncthreads();
    if (tid < GHh) {
      const float* git = gi + (size_t)t * 3 * GHh;
      float r = sigmoidf_(git[tid] + sg[tid]);
      float z = sigmoidf_(git[GHh + tid] + sg[GHh + tid]);
      float n = tanhf(git[2 * GHh + tid] + r * sg[2 * GHh + tid]);
      float hn = (1.f - z) * n + z * sh[tid];
      sh[tid] = hn;
      hout[(size_t)t * Dd + dir * GHh + tid] = hn;
    }
    __syncthreads();
  }
}

// ---------------------------------------------------------------------------
// BiLSTM scan: block 0 = fwd, block 1 = bwd. 1024 threads (32 waves).
// Recurrent weight (1MB) is L2-resident (192MB L2); pre-transposed so the
// per-step read WhhT[k*1024 + tid] is fully lane-coalesced. Gates/state in LDS.
// PyTorch i,f,g,o gate order.
// ---------------------------------------------------------------------------
__global__ void lstm_scan_kernel(const float* __restrict__ giF, const float* __restrict__ giB,
                                 const float* __restrict__ WTf, const float* __restrict__ WTb,
                                 const float* __restrict__ bhhF, const float* __restrict__ bhhB,
                                 float* __restrict__ comb) {
  __shared__ float sh[LHh], sc[LHh], sg[4 * LHh], sb[4 * LHh];
  const int dir = blockIdx.x;
  const float* gi   = dir ? giB  : giF;
  const float* WhhT = dir ? WTb  : WTf;   // [LHh, 4*LHh]
  const float* bhh  = dir ? bhhB : bhhF;
  const int tid = threadIdx.x;  // 0..1023
  sb[tid] = bhh[tid];
  if (tid < LHh) { sh[tid] = 0.f; sc[tid] = 0.f; }
  __syncthreads();
  for (int s = 0; s < Tn; ++s) {
    const int t = dir ? (Tn - 1 - s) : s;
    float acc = sb[tid] + gi[(size_t)t * 4 * LHh + tid];
#pragma unroll 8
    for (int k = 0; k < LHh; ++k) acc += sh[k] * WhhT[(size_t)k * 4 * LHh + tid];
    sg[tid] = acc;
    __syncthreads();
    if (tid < LHh) {
      float i_ = sigmoidf_(sg[tid]);
      float f_ = sigmoidf_(sg[LHh + tid]);
      float g_ = tanhf(sg[2 * LHh + tid]);
      float o_ = sigmoidf_(sg[3 * LHh + tid]);
      float cn = f_ * sc[tid] + i_ * g_;
      float hn = o_ * tanhf(cn);
      sc[tid] = cn; sh[tid] = hn;
      comb[(size_t)t * Ee + dir * LHh + tid] = hn;
    }
    __syncthreads();
  }
}

__global__ void softmax_rows_kernel(const float* __restrict__ logits, float* __restrict__ out) {
  __shared__ float red[128];
  const int row = blockIdx.x;
  const int j = threadIdx.x;  // 128
  float v = (j < NT) ? logits[row * NT + j] : -1e30f;
  red[j] = v; __syncthreads();
  for (int k = 64; k > 0; k >>= 1) { if (j < k) red[j] = fmaxf(red[j], red[j + k]); __syncthreads(); }
  float mx = red[0]; __syncthreads();
  float e = (j < NT) ? expf(v - mx) : 0.f;
  red[j] = e; __syncthreads();
  for (int k = 64; k > 0; k >>= 1) { if (j < k) red[j] += red[j + k]; __syncthreads(); }
  if (j < NT) out[row * NT + j] = e / red[0];
}

// ---------------------------------------------------------------------------
extern "C" void kernel_launch(void* const* d_in, const int* in_sizes, int n_in,
                              void* d_out, int out_size, void* d_ws, size_t ws_size,
                              hipStream_t stream) {
  const float* memory = (const float*)d_in[0];
  const float* u      = (const float*)d_in[1];
  const float* embed  = (const float*)d_in[2];
  const float* W_ff   = (const float*)d_in[3];
  const float* b_ff   = (const float*)d_in[4];
  const float* gWih_f = (const float*)d_in[5];
  const float* gWhh_f = (const float*)d_in[6];
  const float* gbih_f = (const float*)d_in[7];
  const float* gbhh_f = (const float*)d_in[8];
  const float* gWih_b = (const float*)d_in[9];
  const float* gWhh_b = (const float*)d_in[10];
  const float* gbih_b = (const float*)d_in[11];
  const float* gbhh_b = (const float*)d_in[12];
  const float* W_k    = (const float*)d_in[13];
  const float* b_k    = (const float*)d_in[14];
  const float* lWih_f = (const float*)d_in[15];
  const float* lWhh_f = (const float*)d_in[16];
  const float* lbih_f = (const float*)d_in[17];
  const float* lbhh_f = (const float*)d_in[18];
  const float* lWih_b = (const float*)d_in[19];
  const float* lWhh_b = (const float*)d_in[20];
  const float* lbih_b = (const float*)d_in[21];
  const float* lbhh_b = (const float*)d_in[22];
  const float* W_out  = (const float*)d_in[23];
  const float* b_out  = (const float*)d_in[24];
  float* out = (float*)d_out;

  float* ws = (float*)d_ws;
  float* scores = ws;  ws += 512;
  float* p      = ws;  ws += 512;
  float* mbar   = ws;  ws += Tn * Dd;
  float* cat    = ws;  ws += Tn * 2 * Dd;
  float* Gm     = ws;  ws += Tn * Dd;
  float* gi_gf  = ws;  ws += Tn * 3 * GHh;
  float* gi_gb  = ws;  ws += Tn * 3 * GHh;
  float* hgru   = ws;  ws += Tn * Dd;
  float* xk     = ws;  ws += Tn * Dd;
  float* obuf   = ws;  ws += Tn * Ee;
  float* xbuf   = ws;  ws += Tn * Ee;
  float* gi_lf  = ws;  ws += (size_t)Tn * 4 * LHh;
  float* gi_lb  = ws;  ws += (size_t)Tn * 4 * LHh;
  float* WTf    = ws;  ws += 4 * LHh * LHh;
  float* WTb    = ws;  ws += 4 * LHh * LHh;
  float* comb   = ws;  ws += Tn * Ee;
  float* logits = ws;  ws += Tn * NT;

  // Stage 1: attention over memories (bandwidth-bound: 2 passes of 134 MB)
  mem_dot_kernel<<<Mm, 256, 0, stream>>>(memory, u, scores);
  softmax512_kernel<<<1, 512, 0, stream>>>(scores, p);
  mem_wsum_kernel<<<Tn, Dd, 0, stream>>>(memory, p, mbar);
  // ff pulled outside the sum over memories (linearity): G = ff([mbar, u])
  concat_kernel<<<Tn, 256, 0, stream>>>(mbar, u, cat);
  gemm_wmma_bias<<<dim3(Dd / 32, Tn / 64), 128, 0, stream>>>(cat, W_ff, b_ff, Gm, Dd, 2 * Dd);

  // Stage 2: BiGRU — WMMA input projections + sequential LDS scan
  gemm_wmma_bias<<<dim3((3 * GHh) / 32, Tn / 64), 128, 0, stream>>>(Gm, gWih_f, gbih_f, gi_gf, 3 * GHh, Dd);
  gemm_wmma_bias<<<dim3((3 * GHh) / 32, Tn / 64), 128, 0, stream>>>(Gm, gWih_b, gbih_b, gi_gb, 3 * GHh, Dd);
  gru_scan_kernel<<<2, 3 * GHh, 0, stream>>>(gi_gf, gi_gb, gWhh_f, gWhh_b, gbhh_f, gbhh_b, hgru);

  // Stage 3: knowledge encoder
  add_kernel<<<(Tn * Dd + 255) / 256, 256, 0, stream>>>(u, hgru, xk, Tn * Dd);
  gemm_wmma_bias<<<dim3(Ee / 32, Tn / 64), 128, 0, stream>>>(xk, W_k, b_k, obuf, Ee, Dd);

  // Stage 4: BiLSTM — WMMA input projections + L2-resident recurrent scan
  add_kernel<<<(Tn * Ee + 255) / 256, 256, 0, stream>>>(embed, obuf, xbuf, Tn * Ee);
  gemm_wmma_bias<<<dim3((4 * LHh) / 32, Tn / 64), 128, 0, stream>>>(xbuf, lWih_f, lbih_f, gi_lf, 4 * LHh, Ee);
  gemm_wmma_bias<<<dim3((4 * LHh) / 32, Tn / 64), 128, 0, stream>>>(xbuf, lWih_b, lbih_b, gi_lb, 4 * LHh, Ee);
  transpose_kernel<<<(4 * LHh * LHh + 255) / 256, 256, 0, stream>>>(lWhh_f, WTf, 4 * LHh, LHh);
  transpose_kernel<<<(4 * LHh * LHh + 255) / 256, 256, 0, stream>>>(lWhh_b, WTb, 4 * LHh, LHh);
  lstm_scan_kernel<<<2, 4 * LHh, 0, stream>>>(gi_lf, gi_lb, WTf, WTb, lbhh_f, lbhh_b, comb);

  // Stage 5: tagging head
  gemm_wmma_bias<<<dim3((NT + 31) / 32, Tn / 64), 128, 0, stream>>>(comb, W_out, b_out, logits, NT, Ee);
  softmax_rows_kernel<<<Tn, 128, 0, stream>>>(logits, out);
}
